// CausalGraphBuilder_10806137717189
// MI455X (gfx1250) — compile-verified, matrix operand order checked by reading, not run on hardware
//
#include <hip/hip_runtime.h>
#include <hip/hip_bf16.h>

// ---------------------------------------------------------------------------
// CausalGraphBuilder for MI455X (gfx1250), wave32 + WMMA bf16 16x16x32.
//
// Problem: B=128, S=1024, D=512, N=64, DM=128.
//   nodes = gather(state, linspace idx)            (8192 x 512)
//   h  = relu(nodes@W1+b1)@W2+b2                   (8192 x 128)
//   nf = LayerNorm(h)*gamma+beta                   -> output 0
//   pair(i,j) = [nf_i ; nf_j]  (524288 x 256)
//   adjacency = sigmoid(relu(relu(pair@We1+be1)@We2+be2)@We3+be3) * offdiag
//   strengths = tanh(relu(pair@Ws1+bs1)@Ws2+bs2) * offdiag
//
// Factorization: pair@We1 = nf_i@We1[:128] + nf_j@We1[128:], same for Ws1.
// So layer-1 edge GEMMs run at M=8192 instead of M=524288 (per-node, not
// per-pair), cutting total FLOPs ~6x. Only e@We2 remains per-pair.
// ---------------------------------------------------------------------------

typedef __attribute__((ext_vector_type(16))) __bf16 v16bf;
typedef __attribute__((ext_vector_type(8)))  __bf16 v8bf;
typedef __attribute__((ext_vector_type(8)))  float  v8f;

#define BQ   128
#define SS   1024
#define DD   512
#define NN   64
#define DM   128
#define H1   256   // D/2
#define ROWS 8192  // B*N

// WMMA bf16 fragment loader. ISA 16-bit A layout (CDNA5 ISA 7.12.2):
//   lanes 0-15 : M=lane,    VGPR0..3 = K k0..k0+7,  VGPR4..7 = K k0+16..k0+23, k0=0
//   lanes16-31 : M=lane-16, same with k0=8
// B mirrors A with N in place of M (B stored K-major per output column), so
// one loader serves both: `p` points at the 32-wide K window of row m / col n.
__device__ __forceinline__ v16bf load_frag(const __bf16* p, int k0) {
    union { v16bf v; v8bf h[2]; } u;
    u.h[0] = *(const v8bf*)(p + k0);        // K = k0 .. k0+7
    u.h[1] = *(const v8bf*)(p + k0 + 16);   // K = k0+16 .. k0+23
    return u.v;
}

__device__ __forceinline__ v8f wmma_bf16(v16bf a, v16bf b, v8f c) {
    return __builtin_amdgcn_wmma_f32_16x16x32_bf16(false, a, false, b,
                                                   (short)0, c, false, false);
}

// ---------------------------------------------------------------------------
// Kernel 1: per 16-row node tile: gather -> MLP -> LayerNorm -> nf (d_out)
// and factored edge projections ei/ej/si/sj (workspace).
// ---------------------------------------------------------------------------
__global__ __launch_bounds__(256) void k1_nodes(
    const float* __restrict__ state, const float* __restrict__ W1,
    const float* __restrict__ b1,    const float* __restrict__ W2,
    const float* __restrict__ b2,    const float* __restrict__ gamma,
    const float* __restrict__ beta,  const float* __restrict__ We1,
    const float* __restrict__ Ws1,
    float* __restrict__ out_nf, float* __restrict__ eiG, float* __restrict__ ejG,
    float* __restrict__ siG, float* __restrict__ sjG)
{
    __shared__ __bf16 ldsA[16 * DD];    // node tile, [m][k]          16 KB
    __shared__ __bf16 ldsW[32 * H1];    // weight chunk, [n][k32]     16 KB
    __shared__ __bf16 ldsR[16 * H1];    // relu(x@W1) tile / nf tile   8 KB
    __shared__ float  ldsH[16 * DM];    // h tile (fp32 for LN)        8 KB
    __shared__ float  ldsMu[16], ldsRs[16];

    const int t     = threadIdx.x;
    const int lane  = t & 31;
    const int wave  = t >> 5;
    const int mLane = lane & 15;
    const int k0    = (lane >> 4) * 8;
    const int rowBase = blockIdx.x * 16;

    // ---- gather nodes (fp32 -> bf16) ------------------------------------
    for (int u = 0; u < 32; ++u) {
        int l = t + 256 * u;                 // 16*512 = 8192 elements
        int m = l >> 9, c = l & 511;
        int r = rowBase + m;
        int b = r >> 6, i = r & 63;
        int srow = (i * (SS - 1)) / (NN - 1);   // floor(linspace)
        ldsA[m * DD + c] = (__bf16)state[((size_t)b * SS + srow) * DD + c];
    }
    __syncthreads();

    // ---- GEMM1: (16x512) @ W1 (512x256), wave w owns cols 32w..32w+31 ---
    v8f acc0 = {}; v8f acc1 = {};
    for (int kc = 0; kc < 16; ++kc) {
        for (int u = 0; u < 32; ++u) {       // 32x256 chunk, stored [n][k]
            int l = t + 256 * u;
            int kr = l >> 8, n = l & 255;
            ldsW[n * 32 + kr] = (__bf16)W1[(size_t)(kc * 32 + kr) * H1 + n];
        }
        __syncthreads();
        v16bf a = load_frag(&ldsA[mLane * DD + kc * 32], k0);
        v16bf bA = load_frag(&ldsW[(wave * 32 + mLane) * 32], k0);
        acc0 = wmma_bf16(a, bA, acc0);
        v16bf bB = load_frag(&ldsW[(wave * 32 + 16 + mLane) * 32], k0);
        acc1 = wmma_bf16(a, bB, acc1);
        __syncthreads();
    }
    // bias + relu -> r tile (bf16, [m][256])
#pragma unroll
    for (int p = 0; p < 8; ++p) {
        int m = (lane >> 4) * 8 + p;
        int n0 = wave * 32 + mLane;
        int n1 = n0 + 16;
        ldsR[m * H1 + n0] = (__bf16)fmaxf(acc0[p] + b1[n0], 0.f);
        ldsR[m * H1 + n1] = (__bf16)fmaxf(acc1[p] + b1[n1], 0.f);
    }
    __syncthreads();

    // ---- GEMM2: (16x256) @ W2 (256x128), wave w owns cols 16w..16w+15 ---
    v8f acc = {};
    for (int kc = 0; kc < 8; ++kc) {
        for (int u = 0; u < 16; ++u) {       // 32x128 chunk, stored [n][k]
            int l = t + 256 * u;
            int kr = l >> 7, n = l & 127;
            ldsW[n * 32 + kr] = (__bf16)W2[(size_t)(kc * 32 + kr) * DM + n];
        }
        __syncthreads();
        v16bf a = load_frag(&ldsR[mLane * H1 + kc * 32], k0);
        v16bf bA = load_frag(&ldsW[(wave * 16 + mLane) * 32], k0);
        acc = wmma_bf16(a, bA, acc);
        __syncthreads();
    }
#pragma unroll
    for (int p = 0; p < 8; ++p) {
        int m = (lane >> 4) * 8 + p;
        int n = wave * 16 + mLane;
        ldsH[m * DM + n] = acc[p] + b2[n];
    }
    __syncthreads();

    // ---- LayerNorm over DM=128 ------------------------------------------
    if (t < 16) {
        float s = 0.f, ss = 0.f;
        for (int n = 0; n < DM; ++n) { float v = ldsH[t * DM + n]; s += v; ss += v * v; }
        float mu = s * (1.f / DM);
        float var = ss * (1.f / DM) - mu * mu;
        ldsMu[t] = mu;
        ldsRs[t] = rsqrtf(var + 1e-5f);
    }
    __syncthreads();
    for (int u = 0; u < 8; ++u) {            // 16*128 = 2048 elements
        int l = t + 256 * u;
        int m = l >> 7, n = l & 127;
        float v = (ldsH[l] - ldsMu[m]) * ldsRs[m] * gamma[n] + beta[n];
        out_nf[(size_t)(rowBase + m) * DM + n] = v;
        ldsR[m * DM + n] = (__bf16)v;        // nf bf16 tile, [m][128]
    }
    __syncthreads();

    // ---- factored edge projections: nf @ [We1_top|We1_bot|Ws1_top|Ws1_bot]
    // 192 output cols = 12 tile-tasks; wave w -> task w, waves 0-3 also 8+w.
    v8f ec0 = {}; v8f ec1 = {};
    for (int kc = 0; kc < 4; ++kc) {
        for (int u = 0; u < 24; ++u) {       // 32 x 192 combined chunk [c][k]
            int l = t + 256 * u;
            int kr = l / 192, c = l % 192;
            int kg = kc * 32 + kr;
            float wv;
            if (c < 64)       wv = We1[(size_t)kg * 64 + c];
            else if (c < 128) wv = We1[(size_t)(128 + kg) * 64 + (c - 64)];
            else if (c < 160) wv = Ws1[(size_t)kg * 32 + (c - 128)];
            else              wv = Ws1[(size_t)(128 + kg) * 32 + (c - 160)];
            ldsW[c * 32 + kr] = (__bf16)wv;
        }
        __syncthreads();
        v16bf a = load_frag(&ldsR[mLane * DM + kc * 32], k0);
        v16bf bA = load_frag(&ldsW[(wave * 16 + mLane) * 32], k0);
        ec0 = wmma_bf16(a, bA, ec0);
        if (wave < 4) {
            v16bf bB = load_frag(&ldsW[((8 + wave) * 16 + mLane) * 32], k0);
            ec1 = wmma_bf16(a, bB, ec1);
        }
        __syncthreads();
    }
    auto writeTask = [&](int T, const v8f& cc) {
        int cCol = T * 16 + mLane;
        float* dst; int ldc, col;
        if (cCol < 64)       { dst = eiG; ldc = 64; col = cCol; }
        else if (cCol < 128) { dst = ejG; ldc = 64; col = cCol - 64; }
        else if (cCol < 160) { dst = siG; ldc = 32; col = cCol - 128; }
        else                 { dst = sjG; ldc = 32; col = cCol - 160; }
#pragma unroll
        for (int p = 0; p < 8; ++p) {
            int m = (lane >> 4) * 8 + p;
            dst[(size_t)(rowBase + m) * ldc + col] = cc[p];
        }
    };
    writeTask(wave, ec0);
    if (wave < 4) writeTask(8 + wave, ec1);
}

// ---------------------------------------------------------------------------
// Kernel 2: per (batch, 16x16 pair tile): e=relu(ei+ej+be1) built directly
// into WMMA A-fragments, @We2 via WMMA, then tiny 32-wide dots + activations.
// ---------------------------------------------------------------------------
__global__ __launch_bounds__(256) void k2_pairs(
    const float* __restrict__ eiG, const float* __restrict__ ejG,
    const float* __restrict__ siG, const float* __restrict__ sjG,
    const float* __restrict__ be1, const float* __restrict__ We2,
    const float* __restrict__ be2, const float* __restrict__ We3,
    const float* __restrict__ be3, const float* __restrict__ bs1,
    const float* __restrict__ Ws2, const float* __restrict__ bs2,
    float* __restrict__ outAdj, float* __restrict__ outStr)
{
    __shared__ float  eiT[16 * 64], ejT[16 * 64];
    __shared__ float  siT[16 * 32], sjT[16 * 32];
    __shared__ __bf16 we2T[32 * 64];          // [n][k] bf16
    __shared__ float  e2buf[16 * 16 * 32];    // relu(e@We2+be2), 32 KB
    __shared__ float  sbe1[64], sbe2[32], sWe3[32], sbs1[32], sWs2[32];

    const int t = threadIdx.x, lane = t & 31, wave = t >> 5;
    const int blk = blockIdx.x;
    const int b  = blk >> 4;
    const int it = (blk >> 2) & 3;
    const int jt = blk & 3;
    const int iBase = it * 16, jBase = jt * 16;
    const size_t rowI = (size_t)b * NN + iBase;
    const size_t rowJ = (size_t)b * NN + jBase;

    for (int u = 0; u < 4; ++u) {             // 16x64 each
        int l = t + 256 * u; int m = l >> 6, c = l & 63;
        eiT[l] = eiG[(rowI + m) * 64 + c];
        ejT[l] = ejG[(rowJ + m) * 64 + c];
    }
    for (int u = 0; u < 2; ++u) {             // 16x32 each
        int l = t + 256 * u; int m = l >> 5, c = l & 31;
        siT[l] = siG[(rowI + m) * 32 + c];
        sjT[l] = sjG[(rowJ + m) * 32 + c];
    }
    for (int u = 0; u < 8; ++u) {             // We2: 64x32 -> [n][k]
        int l = t + 256 * u; int k = l >> 5, n = l & 31;
        we2T[n * 64 + k] = (__bf16)We2[k * 32 + n];
    }
    if (t < 64) sbe1[t] = be1[t];
    if (t < 32) { sbe2[t] = be2[t]; sWe3[t] = We3[t]; sbs1[t] = bs1[t]; sWs2[t] = Ws2[t]; }
    __syncthreads();

    const int mLane = lane & 15, k0 = (lane >> 4) * 8;
    for (int g = 0; g < 2; ++g) {             // wave handles i-rows w and w+8
        int iloc = wave + g * 8;
        const float* eiRow = &eiT[iloc * 64];
        v8f c0 = {}; v8f c1 = {};
        for (int kc = 0; kc < 2; ++kc) {
            v16bf a;                          // A row m = j-index = mLane
#pragma unroll
            for (int x = 0; x < 16; ++x) {
                int k = kc * 32 + k0 + (x & 7) + ((x >> 3) << 4);
                float v = eiRow[k] + ejT[mLane * 64 + k] + sbe1[k];
                a[x] = (__bf16)fmaxf(v, 0.f);
            }
            v16bf b0 = load_frag(&we2T[mLane * 64 + kc * 32], k0);
            c0 = wmma_bf16(a, b0, c0);
            v16bf b1f = load_frag(&we2T[(16 + mLane) * 64 + kc * 32], k0);
            c1 = wmma_bf16(a, b1f, c1);
        }
#pragma unroll
        for (int p = 0; p < 8; ++p) {
            int j = (lane >> 4) * 8 + p;
            int n0 = mLane, n1 = 16 + mLane;
            e2buf[(iloc * 16 + j) * 32 + n0] = fmaxf(c0[p] + sbe2[n0], 0.f);
            e2buf[(iloc * 16 + j) * 32 + n1] = fmaxf(c1[p] + sbe2[n1], 0.f);
        }
    }
    __syncthreads();

    // one pair per thread: final dots + sigmoid/tanh + diagonal mask
    {
        int iloc = t >> 4, jloc = t & 15;
        const float* e2 = &e2buf[(iloc * 16 + jloc) * 32];
        float da = 0.f, ds = 0.f;
#pragma unroll
        for (int k = 0; k < 32; ++k) {
            da += e2[k] * sWe3[k];
            float sv = fmaxf(siT[iloc * 32 + k] + sjT[jloc * 32 + k] + sbs1[k], 0.f);
            ds += sv * sWs2[k];
        }
        float adj = 1.f / (1.f + __expf(-(da + be3[0])));
        float stv = tanhf(ds + bs2[0]);
        int gi = iBase + iloc, gj = jBase + jloc;
        if (gi == gj) { adj = 0.f; stv = 0.f; }
        size_t o = ((size_t)b * NN + gi) * NN + gj;
        outAdj[o] = adj;
        outStr[o] = stv;
    }
}

// ---------------------------------------------------------------------------
extern "C" void kernel_launch(void* const* d_in, const int* in_sizes, int n_in,
                              void* d_out, int out_size, void* d_ws, size_t ws_size,
                              hipStream_t stream) {
    (void)in_sizes; (void)n_in; (void)out_size; (void)ws_size;
    const float* state = (const float*)d_in[0];
    const float* W1    = (const float*)d_in[1];
    const float* b1    = (const float*)d_in[2];
    const float* W2    = (const float*)d_in[3];
    const float* b2    = (const float*)d_in[4];
    const float* gamma = (const float*)d_in[5];
    const float* beta  = (const float*)d_in[6];
    const float* We1   = (const float*)d_in[7];
    const float* be1   = (const float*)d_in[8];
    const float* We2   = (const float*)d_in[9];
    const float* be2   = (const float*)d_in[10];
    const float* We3   = (const float*)d_in[11];
    const float* be3   = (const float*)d_in[12];
    const float* Ws1   = (const float*)d_in[13];
    const float* bs1   = (const float*)d_in[14];
    const float* Ws2   = (const float*)d_in[15];
    const float* bs2   = (const float*)d_in[16];

    float* out_nf  = (float*)d_out;
    float* out_adj = out_nf + (size_t)ROWS * DM;        // 128*64*128
    float* out_str = out_adj + (size_t)BQ * NN * NN;    // 128*64*64

    float* ei = (float*)d_ws;                 // [8192][64]
    float* ej = ei + (size_t)ROWS * 64;       // [8192][64]
    float* si = ej + (size_t)ROWS * 64;       // [8192][32]
    float* sj = si + (size_t)ROWS * 32;       // [8192][32]   total 6 MB

    k1_nodes<<<ROWS / 16, 256, 0, stream>>>(state, W1, b1, W2, b2, gamma, beta,
                                            We1, Ws1, out_nf, ei, ej, si, sj);
    k2_pairs<<<BQ * 16, 256, 0, stream>>>(ei, ej, si, sj, be1, We2, be2, We3,
                                          be3, bs1, Ws2, bs2, out_adj, out_str);
}
